// BatchMeshDeformationBlock_28269474742811
// MI455X (gfx1250) — matrix operand order repaired and checked by reading, not compiled, not run on hardware
//
#include <hip/hip_runtime.h>

#define B_ 32
#define V_ 2562
#define H_ 192
#define CIN_ 966
#define PD_ 963
#define MTOT (B_*V_)   /* 81984 */
#define MAXN 16
#define EPS_ 1e-5f

typedef __attribute__((ext_vector_type(16))) __bf16 bf16x16;
typedef __attribute__((ext_vector_type(8)))  float  floatx8;

union FragAB { bf16x16 v; unsigned int u[8]; };

__device__ __forceinline__ unsigned short f2bf(float f) {
  union { float f; unsigned int u; } c; c.f = f;
  unsigned int u = c.u;
  u += 0x7FFFu + ((u >> 16) & 1u);          // round-to-nearest-even
  return (unsigned short)(u >> 16);
}

// ---------------------------------------------------------------------------
// Extract sparse adjacency (<=7 nnz/row) + row-sum degree from dense adj.
// ---------------------------------------------------------------------------
__global__ __launch_bounds__(128) void build_graph(
    const float* __restrict__ adj, int* __restrict__ nbr_idx,
    float* __restrict__ nbr_val, int* __restrict__ nbr_cnt,
    float* __restrict__ norm)
{
  const int u = blockIdx.x;
  const int tid = threadIdx.x;
  __shared__ int cnt;
  __shared__ float red[128];
  if (tid == 0) cnt = 0;
  __syncthreads();
  float s = 0.0f;
  for (int v = tid; v < V_; v += 128) {
    float a = adj[u*V_ + v];
    if (a != 0.0f) {
      s += a;
      int slot = atomicAdd(&cnt, 1);
      if (slot < MAXN) { nbr_idx[u*MAXN + slot] = v; nbr_val[u*MAXN + slot] = a; }
    }
  }
  red[tid] = s; __syncthreads();
  for (int st = 64; st > 0; st >>= 1) {
    if (tid < st) red[tid] += red[tid + st];
    __syncthreads();
  }
  if (tid == 0) { nbr_cnt[u] = cnt < MAXN ? cnt : MAXN; norm[u] = red[0]; }
}

// feats0 = concat(features, pooled)[:, :, :192]
__global__ __launch_bounds__(256) void init_feats(
    const float* __restrict__ features, const float* __restrict__ pooled,
    float* __restrict__ feats)
{
  int idx = blockIdx.x*256 + threadIdx.x;
  if (idx >= MTOT*H_) return;
  int row = idx / H_, c = idx % H_;
  feats[idx] = (c < 3) ? features[row*3 + c] : pooled[row*PD_ + (c - 3)];
}

// ---------------------------------------------------------------------------
// Tiled bf16 WMMA GEMM: support[M,192] = X[M,K] @ W[K,192]
//   channels 0..63 pre-divided by per-vertex degree (support[...,:64]/norm)
// mode 0: A = X[row*K + k] ; mode 1 (gc1): A = concat(features, pooled)
// ---------------------------------------------------------------------------
__global__ __launch_bounds__(256) void gemm_gcn(
    const float* __restrict__ X,
    const float* __restrict__ featIn,
    const float* __restrict__ pooled,
    const float* __restrict__ W,
    const float* __restrict__ norm,
    float* __restrict__ support,
    int K, int mode)
{
  __shared__ unsigned short sA[128*34];   // A tile, bf16, stride 34 (conflict-free)
  __shared__ unsigned short sB[192*34];   // W tile transposed [n][k]
  const int tid  = threadIdx.x;
  const int wave = tid >> 5;
  const int lane = tid & 31;
  const int half = lane >> 4;
  const int ln   = lane & 15;
  const int m0   = blockIdx.x * 128;

  floatx8 acc[12] = {};

  for (int k0 = 0; k0 < K; k0 += 32) {
    __syncthreads();
    // stage A (128x32 fp32 -> bf16)
    for (int i = tid; i < 128*32; i += 256) {
      int r = i >> 5, kk = i & 31;
      int row = m0 + r, gk = k0 + kk;
      float v = 0.0f;
      if (row < MTOT && gk < K) {
        if (mode == 0) v = X[row*K + gk];
        else           v = (gk < 3) ? featIn[row*3 + gk]
                                    : pooled[row*PD_ + (gk - 3)];
      }
      sA[r*34 + kk] = f2bf(v);
    }
    // stage W (32x192 fp32 -> bf16, transposed into [n][k])
    for (int i = tid; i < 32*192; i += 256) {
      int kk = i / 192, n = i % 192;
      int gk = k0 + kk;
      float v = (gk < K) ? W[gk*192 + n] : 0.0f;
      sB[n*34 + kk] = f2bf(v);
    }
    __syncthreads();

    // A fragment (16x32 bf16, ISA 7.12.2 layout)
    FragAB a;
    const int mrow = wave*16 + ln;
    #pragma unroll
    for (int vi = 0; vi < 8; ++vi) {
      int kb = (vi < 4) ? (2*vi) : (16 + 2*(vi - 4));
      int k  = kb + 8*half;
      a.u[vi] = *(const unsigned int*)&sA[mrow*34 + k];
    }
    #pragma unroll
    for (int j = 0; j < 12; ++j) {
      FragAB bf;
      #pragma unroll
      for (int vi = 0; vi < 8; ++vi) {
        int k = ((vi < 4) ? (2*vi) : (16 + 2*(vi - 4))) + 8*half;  // B 32x16: same K order as A (lanes 0-15 K=0..7,16..23; 16-31 K=8..15,24..31)
        bf.u[vi] = *(const unsigned int*)&sB[(16*j + ln)*34 + k];
      }
      acc[j] = __builtin_amdgcn_wmma_f32_16x16x32_bf16(
          false, a.v, false, bf.v, (short)0, acc[j], false, false);
    }
  }

  // epilogue: C layout -> row = base + r + 8*half, col = 16j + (lane&15)
  #pragma unroll
  for (int r = 0; r < 8; ++r) {
    int row = m0 + wave*16 + 8*half + r;
    if (row < MTOT) {
      float inv = 1.0f / norm[row % V_];
      #pragma unroll
      for (int j = 0; j < 12; ++j) {
        float val = acc[j][r];
        if (j < 4) val *= inv;                  // cols 0..63: support/norm
        support[row*192 + 16*j + ln] = val;
      }
    }
  }
}

// ---------------------------------------------------------------------------
// Graph aggregation + bias + per-vertex BN statistics. One block per vertex.
// y[b,u,c] = (c<128 ? support[b,u,64+c] : sum_nbr support[b,v,c-128]) + bias[c]
// ---------------------------------------------------------------------------
__global__ __launch_bounds__(256) void aggregate_stats(
    const float* __restrict__ support, const int* __restrict__ nbr_idx,
    const float* __restrict__ nbr_val, const int* __restrict__ nbr_cnt,
    const float* __restrict__ bias, float* __restrict__ y,
    float* __restrict__ meanA, float* __restrict__ rstdA)
{
  const int u = blockIdx.x;
  const int tid = threadIdx.x;
  __shared__ int s_idx[MAXN];
  __shared__ float s_val[MAXN];
  __shared__ int s_cnt;
  __shared__ float red[256], red2[256];
  if (tid == 0) s_cnt = nbr_cnt[u];
  if (tid < MAXN) { s_idx[tid] = nbr_idx[u*MAXN + tid]; s_val[tid] = nbr_val[u*MAXN + tid]; }
  __syncthreads();
  const int cnt = s_cnt;
  float sum = 0.0f, sq = 0.0f;
  for (int t = tid; t < B_*H_; t += 256) {
    int b = t / H_, c = t % H_;
    float val;
    if (c < 128) {
      val = support[(b*V_ + u)*H_ + 64 + c];
    } else {
      int d = c - 128;
      float s = 0.0f;
      for (int j = 0; j < cnt; ++j)
        s += s_val[j] * support[(b*V_ + s_idx[j])*H_ + d];
      val = s;
    }
    val += bias[c];
    y[(b*V_ + u)*H_ + c] = val;
    sum += val; sq += val*val;
  }
  red[tid] = sum; red2[tid] = sq;
  __syncthreads();
  for (int st = 128; st > 0; st >>= 1) {
    if (tid < st) { red[tid] += red[tid+st]; red2[tid] += red2[tid+st]; }
    __syncthreads();
  }
  if (tid == 0) {
    float m   = red[0]  / (float)(B_*H_);
    float var = red2[0] / (float)(B_*H_) - m*m;
    meanA[u] = m;
    rstdA[u] = rsqrtf(var + EPS_);
  }
}

// BN + ReLU (+ optional residual average)
__global__ __launch_bounds__(256) void bn_relu_kernel(
    const float* __restrict__ y, const float* __restrict__ meanA,
    const float* __restrict__ rstdA, const float* __restrict__ g,
    const float* __restrict__ bt, const float* __restrict__ resid,
    float* __restrict__ out, int mode)
{
  int idx = blockIdx.x*256 + threadIdx.x;
  if (idx >= MTOT*H_) return;
  int row = idx / H_;
  int v = row % V_;
  float x = g[v]*(y[idx] - meanA[v])*rstdA[v] + bt[v];
  x = fmaxf(x, 0.0f);
  out[idx] = mode ? 0.5f*(resid[idx] + x) : x;
}

// gc15: support3 = feats @ W15 (192x3), ch0/ch1 pre-divided by degree
__global__ __launch_bounds__(256) void gemm15(
    const float* __restrict__ feats, const float* __restrict__ W15,
    const float* __restrict__ norm, float* __restrict__ sup3)
{
  int row = blockIdx.x*256 + threadIdx.x;
  if (row >= MTOT) return;
  const float* xr = feats + row*H_;
  float a0 = 0.f, a1 = 0.f, a2 = 0.f;
  for (int k = 0; k < H_; ++k) {
    float x = xr[k];
    a0 = fmaf(x, W15[k*3+0], a0);
    a1 = fmaf(x, W15[k*3+1], a1);
    a2 = fmaf(x, W15[k*3+2], a2);
  }
  float inv = 1.0f / norm[row % V_];
  sup3[row*3+0] = a0*inv;
  sup3[row*3+1] = a1*inv;
  sup3[row*3+2] = a2;
}

// coords = concat(support3[...,2:], side) + b15
__global__ __launch_bounds__(256) void coords_kernel(
    const float* __restrict__ sup3,
    const int* __restrict__ nbr_idx, const float* __restrict__ nbr_val,
    const int* __restrict__ nbr_cnt, const float* __restrict__ b15,
    float* __restrict__ coords)
{
  int row = blockIdx.x*256 + threadIdx.x;
  if (row >= MTOT) return;
  int u = row % V_; int b = row / V_;
  int cnt = nbr_cnt[u];
  float s0 = 0.f, s1 = 0.f;
  for (int j = 0; j < cnt; ++j) {
    int v = nbr_idx[u*MAXN + j];
    float a = nbr_val[u*MAXN + j];
    const float* p = sup3 + (b*V_ + v)*3;
    s0 += a * p[0];
    s1 += a * p[1];
  }
  coords[row*3 + 0] = sup3[row*3 + 2] + b15[0];
  coords[row*3 + 1] = s0 + b15[1];
  coords[row*3 + 2] = s1 + b15[2];
}

// ---------------------------------------------------------------------------
extern "C" void kernel_launch(void* const* d_in, const int* in_sizes, int n_in,
                              void* d_out, int out_size, void* d_ws, size_t ws_size,
                              hipStream_t stream)
{
  const float* features = (const float*)d_in[0];
  const float* pooled   = (const float*)d_in[1];
  const float* adj      = (const float*)d_in[2];
  const float* W1       = (const float*)d_in[3];
  const float* b1       = (const float*)d_in[4];
  const float* Wm       = (const float*)d_in[5];
  const float* bm       = (const float*)d_in[6];
  const float* W15      = (const float*)d_in[7];
  const float* b15      = (const float*)d_in[8];
  const float* gamma    = (const float*)d_in[9];
  const float* beta     = (const float*)d_in[10];

  float* feats  = (float*)d_out;              // [MTOT, 192] (= first output)
  float* coords = feats + (size_t)MTOT*H_;    // [MTOT, 3]

  float* ws      = (float*)d_ws;
  float* support = ws;                           // MTOT*192
  float* ybuf    = support + (size_t)MTOT*H_;    // MTOT*192
  float* xbuf    = ybuf    + (size_t)MTOT*H_;    // MTOT*192
  float* sup3    = xbuf    + (size_t)MTOT*H_;    // MTOT*3
  float* normp   = sup3    + (size_t)MTOT*3;     // V
  float* meanp   = normp + V_;
  float* rstdp   = meanp + V_;
  float* nbrval  = rstdp + V_;                   // V*MAXN
  int*   nbridx  = (int*)(nbrval + V_*MAXN);     // V*MAXN
  int*   nbrcnt  = nbridx + V_*MAXN;             // V

  const int gemmGrid = (MTOT + 127)/128;
  const int ewGrid   = (MTOT*H_ + 255)/256;
  const int rowGrid  = (MTOT + 255)/256;

  build_graph<<<V_, 128, 0, stream>>>(adj, nbridx, nbrval, nbrcnt, normp);
  init_feats<<<ewGrid, 256, 0, stream>>>(features, pooled, feats);

  // gc1: 966 -> 192
  gemm_gcn<<<gemmGrid, 256, 0, stream>>>(nullptr, features, pooled, W1, normp, support, CIN_, 1);
  aggregate_stats<<<V_, 256, 0, stream>>>(support, nbridx, nbrval, nbrcnt, b1, ybuf, meanp, rstdp);
  bn_relu_kernel<<<ewGrid, 256, 0, stream>>>(ybuf, meanp, rstdp, gamma + 0*V_, beta + 0*V_, nullptr, xbuf, 0);

  // gc2 (Wm[0]): feats = (feats + x)/2
  gemm_gcn<<<gemmGrid, 256, 0, stream>>>(xbuf, nullptr, nullptr, Wm + 0*H_*H_, normp, support, H_, 0);
  aggregate_stats<<<V_, 256, 0, stream>>>(support, nbridx, nbrval, nbrcnt, bm + 0*H_, ybuf, meanp, rstdp);
  bn_relu_kernel<<<ewGrid, 256, 0, stream>>>(ybuf, meanp, rstdp, gamma + 1*V_, beta + 1*V_, feats, feats, 1);

  // five residual pairs: (gc3,gc4) .. (gc11,gc12)
  for (int i = 0; i < 5; ++i) {
    int lA = 2*i + 1, lB = 2*i + 2;
    gemm_gcn<<<gemmGrid, 256, 0, stream>>>(feats, nullptr, nullptr, Wm + lA*H_*H_, normp, support, H_, 0);
    aggregate_stats<<<V_, 256, 0, stream>>>(support, nbridx, nbrval, nbrcnt, bm + lA*H_, ybuf, meanp, rstdp);
    bn_relu_kernel<<<ewGrid, 256, 0, stream>>>(ybuf, meanp, rstdp, gamma + (2*i+2)*V_, beta + (2*i+2)*V_, nullptr, xbuf, 0);

    gemm_gcn<<<gemmGrid, 256, 0, stream>>>(xbuf, nullptr, nullptr, Wm + lB*H_*H_, normp, support, H_, 0);
    aggregate_stats<<<V_, 256, 0, stream>>>(support, nbridx, nbrval, nbrcnt, bm + lB*H_, ybuf, meanp, rstdp);
    bn_relu_kernel<<<ewGrid, 256, 0, stream>>>(ybuf, meanp, rstdp, gamma + (2*i+3)*V_, beta + (2*i+3)*V_, feats, feats, 1);
  }

  // gc13 (Wm[11])
  gemm_gcn<<<gemmGrid, 256, 0, stream>>>(feats, nullptr, nullptr, Wm + 11*H_*H_, normp, support, H_, 0);
  aggregate_stats<<<V_, 256, 0, stream>>>(support, nbridx, nbrval, nbrcnt, bm + 11*H_, ybuf, meanp, rstdp);
  bn_relu_kernel<<<ewGrid, 256, 0, stream>>>(ybuf, meanp, rstdp, gamma + 12*V_, beta + 12*V_, feats, feats, 1);

  // gc15: coords
  gemm15<<<rowGrid, 256, 0, stream>>>(feats, W15, normp, sup3);
  coords_kernel<<<rowGrid, 256, 0, stream>>>(sup3, nbridx, nbrval, nbrcnt, b15, coords);
}
